// TritonAttention_57793079935636
// MI455X (gfx1250) — matrix operand order repaired and checked by reading, not compiled, not run on hardware
//
#include <hip/hip_runtime.h>
#include <hip/hip_bf16.h>

// ---------------------------------------------------------------------------
// CDNA5 (gfx1250) wave32 WMMA attention block:
//   out = Attention(RoPE(x@Wq), RoPE(x@Wk), x@Wv) @ Wo
// B=2, S=2048, DIM=2048, NH=32, NKV=8, HD=64
// - v_wmma_f32_16x16x32_f16 for all matmuls (f32 accumulate, f16 operands)
// - GLOBAL_LOAD_ASYNC_TO_LDS_B128 (inline asm, ASYNCcnt) for Q/K tile staging
// - packed v2h transpose stores for V (ds_store_b32 instead of b16 scatter)
// - fully unrolled register-array access (no v_movrel), global_prefetch_b8
// ---------------------------------------------------------------------------

typedef __attribute__((ext_vector_type(16))) _Float16 v16h;
typedef __attribute__((ext_vector_type(8)))  _Float16 v8h;
typedef __attribute__((ext_vector_type(4)))  _Float16 v4h;
typedef __attribute__((ext_vector_type(2)))  _Float16 v2h;
typedef __attribute__((ext_vector_type(8)))  float    v8f;
typedef __attribute__((ext_vector_type(4)))  float    v4f;

#define WMMA_F32_16x16x32_F16(A, Bv, Cv) \
    __builtin_amdgcn_wmma_f32_16x16x32_f16(false, (A), false, (Bv), (short)0, (Cv), false, false)

// ---- CDNA5 async global->LDS copy (ASYNCcnt) ------------------------------
// global_load_async_to_lds_b128: VDST = per-lane LDS byte offset (VGPR),
// VADDR = 64-bit global address, SADDR = off. Flat->LDS mapping uses
// addr[31:0], so truncating the generic shared pointer yields the LDS offset.
__device__ __forceinline__ void async_g2l_b128(void* lds, const void* gptr) {
    unsigned int loff = (unsigned int)(uintptr_t)lds;
    asm volatile("global_load_async_to_lds_b128 %0, %1, off"
                 :: "v"(loff), "v"(gptr) : "memory");
}
__device__ __forceinline__ void wait_async_zero() {
    asm volatile("s_wait_asynccnt 0" ::: "memory");
}

// ---- fragment loaders (ISA 7.12.2 wave32 layouts) -------------------------

// A fragment: 16(M) x 32(K) f16, row-major LDS tile, row pitch in halves.
__device__ __forceinline__ v16h load_a_frag(const _Float16* base, int pitch) {
    int lane = threadIdx.x & 31;
    int m    = lane & 15;
    int kb   = (lane & 16) ? 8 : 0;
    const _Float16* p = base + m * pitch + kb;
    v8h lo = *(const v8h*)(p);        // K = kb .. kb+7
    v8h hi = *(const v8h*)(p + 16);   // K = kb+16 .. kb+23
    return __builtin_shufflevector(lo, hi, 0,1,2,3,4,5,6,7,8,9,10,11,12,13,14,15);
}

// B fragment: 32(K) x 16(N) f16, from an N-major LDS tile (tile[n][k]).
__device__ __forceinline__ v16h load_b_frag(const _Float16* base, int pitch) {
    int lane = threadIdx.x & 31;
    int n    = lane & 15;
    int kb   = (lane & 16) ? 8 : 0;
    const _Float16* p = base + n * pitch + kb;
    v8h lo = *(const v8h*)(p);
    v8h hi = *(const v8h*)(p + 16);
    return __builtin_shufflevector(lo, hi, 0,1,2,3,4,5,6,7,8,9,10,11,12,13,14,15);
}

// D/C 16x16 f32: VGPR r -> row r (lanes 0-15) / row r+8 (lanes 16-31); col = lane&15.

__device__ __forceinline__ v4h load4_h(const float* p) {
    v4f v = *(const v4f*)p;
    v4h r = { (_Float16)v[0], (_Float16)v[1], (_Float16)v[2], (_Float16)v[3] };
    return r;
}
__device__ __forceinline__ v4h load4_h(const _Float16* p) { return *(const v4h*)p; }
__device__ __forceinline__ void st_out(_Float16* p, float v) { *p = (_Float16)v; }
__device__ __forceinline__ void st_out(float* p, float v)    { *p = v; }

// ---------------------------------------------------------------------------
// Generic tiled WMMA GEMM:  C[M,N] = A[M,K] @ W[K,N]
// Block tile 128x64, BK=32, 256 threads (8 waves: 4x2), wave tile 32x32.
// ---------------------------------------------------------------------------
#define BM 128
#define BN 64
#define BK 32
#define APAD 8

template <typename AT, typename OT>
__global__ __launch_bounds__(256) void gemm_wmma(const AT* __restrict__ A,
                                                 const float* __restrict__ W,
                                                 OT* __restrict__ C,
                                                 int M, int N, int K) {
    __shared__ __align__(32) _Float16 As[BM][BK + APAD];   // row-major (M,K)
    __shared__ __align__(32) _Float16 Bs[BN][BK + APAD];   // N-major  (N,K)

    const int tid  = threadIdx.x;
    const int wave = tid >> 5;
    const int wm   = wave >> 1;      // 0..3
    const int wn   = wave & 1;       // 0..1
    const int bm   = blockIdx.y * BM;
    const int bn   = blockIdx.x * BN;

    v8f acc[2][2] = {};

    for (int kb = 0; kb < K; kb += BK) {
        // stage A tile: 16B vector loads, f16 convert, 8B LDS stores
        #pragma unroll
        for (int it = 0; it < (BM * BK) / (256 * 4); ++it) {      // 4 iters
            int idx = tid + it * 256;
            int r = idx >> 3;                  // (idx*4)/32
            int c = (idx & 7) * 4;
            *(v4h*)&As[r][c] = load4_h(&A[(size_t)(bm + r) * K + kb + c]);
        }
        // stage W tile transposed -> N-major (16B global loads)
        #pragma unroll
        for (int it = 0; it < (BN * BK) / (256 * 4); ++it) {      // 2 iters
            int idx = tid + it * 256;
            int c = idx >> 4;                  // k in tile
            int n = (idx & 15) * 4;
            v4f w = *(const v4f*)&W[(size_t)(kb + c) * N + bn + n];
            Bs[n + 0][c] = (_Float16)w[0];
            Bs[n + 1][c] = (_Float16)w[1];
            Bs[n + 2][c] = (_Float16)w[2];
            Bs[n + 3][c] = (_Float16)w[3];
        }
        // prefetch next K-tile while this one computes (global_prefetch_b8)
        if (kb + BK < K) {
            int pr = tid >> 1, pc = (tid & 1) * 16;
            __builtin_prefetch(&A[(size_t)(bm + pr) * K + kb + BK + pc], 0, 3);
            __builtin_prefetch(&W[(size_t)(kb + BK + (tid & 31)) * N + bn + (tid >> 5) * 8], 0, 3);
        }
        __syncthreads();

        v16h a0 = load_a_frag(&As[wm * 32 +  0][0], BK + APAD);
        v16h a1 = load_a_frag(&As[wm * 32 + 16][0], BK + APAD);
        v16h b0 = load_b_frag(&Bs[wn * 32 +  0][0], BK + APAD);
        v16h b1 = load_b_frag(&Bs[wn * 32 + 16][0], BK + APAD);

        acc[0][0] = WMMA_F32_16x16x32_F16(a0, b0, acc[0][0]);
        acc[0][1] = WMMA_F32_16x16x32_F16(a0, b1, acc[0][1]);
        acc[1][0] = WMMA_F32_16x16x32_F16(a1, b0, acc[1][0]);
        acc[1][1] = WMMA_F32_16x16x32_F16(a1, b1, acc[1][1]);

        __syncthreads();
    }

    const int lane = tid & 31;
    const int nloc = lane & 15;
    const int mhi  = (lane & 16) ? 8 : 0;
    #pragma unroll
    for (int i = 0; i < 2; ++i)
        #pragma unroll
        for (int j = 0; j < 2; ++j)
            #pragma unroll
            for (int r = 0; r < 8; ++r) {
                int row = bm + wm * 32 + i * 16 + mhi + r;
                int col = bn + wn * 32 + j * 16 + nloc;
                st_out(C + (size_t)row * N + col, acc[i][j][r]);
            }
}

// ---------------------------------------------------------------------------
// RoPE (in place on f16 Q or K): pairs (2i,2i+1) within head dim.
// ---------------------------------------------------------------------------
__global__ __launch_bounds__(256) void rope_kernel(_Float16* __restrict__ t,
                                                   const float* __restrict__ cosb,
                                                   const float* __restrict__ sinb,
                                                   int nheads, int total) {
    int idx = blockIdx.x * blockDim.x + threadIdx.x;
    if (idx >= total) return;
    int i   = idx & 31;              // pair index (HD/2 = 32)
    int tmp = idx >> 5;
    int h   = tmp % nheads;
    int bs  = tmp / nheads;          // b*S + s
    int s   = bs & 2047;             // S = 2048
    size_t base = ((size_t)bs * nheads + h) * 64 + 2 * i;
    float c  = cosb[s * 32 + i];
    float sn = sinb[s * 32 + i];
    v2h tv = *(v2h*)&t[base];
    float t0 = (float)tv[0], t1 = (float)tv[1];
    v2h o = { (_Float16)(t0 * c - t1 * sn), (_Float16)(t0 * sn + t1 * c) };
    *(v2h*)&t[base] = o;
}

// ---------------------------------------------------------------------------
// Flash attention (causal, GQA 4:1). One block per (qtile=64, head, batch).
// 4 waves; wave w owns query rows [w*16, w*16+16). Online softmax in f32.
// Q and K tiles staged with async global->LDS b128; V transposed via packed
// v2h stores (two adjacent keys per ds_store_b32).
// ---------------------------------------------------------------------------
__global__ __launch_bounds__(128) void flash_attn(const _Float16* __restrict__ Qh,
                                                  const _Float16* __restrict__ Kh,
                                                  const _Float16* __restrict__ Vh,
                                                  _Float16* __restrict__ Oh) {
    const int S = 2048, NH = 32, NKV = 8, HD = 64;
    const int QW = NH * HD;    // 2048
    const int KW = NKV * HD;   // 512

    __shared__ __align__(32) _Float16 Qs[64][72];       // queries x hd (row-major)
    __shared__ __align__(32) _Float16 Ks[64][72];       // keys x hd  (N-major for Q.K^T)
    __shared__ __align__(32) _Float16 Vt[64][72];       // hd x keys  (N-major for P.V)
    __shared__ __align__(32) _Float16 Ps[4][16][72];    // per-wave P strip

    const int qb   = blockIdx.x;
    const int h    = blockIdx.y;
    const int b    = blockIdx.z;
    const int kvh  = h >> 2;                 // n_rep = 4
    const int tid  = threadIdx.x;
    const int wave = tid >> 5;
    const int lane = tid & 31;
    const int q0   = qb * 64;
    const int wq   = wave * 16;
    const int nloc = lane & 15;
    const int mhi  = (lane & 16) ? 8 : 0;
    const float scale = 0.125f;              // 1/sqrt(64)

    // stage Q tile once: async global -> LDS, 16B per lane
    #pragma unroll
    for (int it = 0; it < 4; ++it) {
        int idx = tid + it * 128;
        int r = idx >> 3;
        int d = (idx & 7) * 8;
        async_g2l_b128(&Qs[r][d], &Qh[(size_t)(b * S + q0 + r) * QW + h * HD + d]);
    }
    wait_async_zero();
    __syncthreads();

    v16h aq0 = load_a_frag(&Qs[wq][0],  72);
    v16h aq1 = load_a_frag(&Qs[wq][32], 72);

    v8f   acc[4] = {};
    float m_s[8], l_s[8];
    #pragma unroll
    for (int r = 0; r < 8; ++r) { m_s[r] = -1e30f; l_s[r] = 0.0f; }

    for (int kb = 0; kb <= qb; ++kb) {
        const int k0 = kb * 64;
        // K tile: async global -> LDS (row-major == N-major for Q.K^T)
        #pragma unroll
        for (int it = 0; it < 4; ++it) {
            int idx = tid + it * 128;
            int r = idx >> 3;
            int d = (idx & 7) * 8;
            async_g2l_b128(&Ks[r][d],
                           &Kh[(size_t)(b * S + k0 + r) * KW + kvh * HD + d]);
        }
        // V tile: transpose via key-pairs -> packed v2h (ds_store_b32)
        #pragma unroll
        for (int it = 0; it < 2; ++it) {
            int idx = tid + it * 128;          // 0..255
            int rp  = idx >> 3;                // key pair 0..31
            int d   = (idx & 7) * 8;
            size_t off0 = (size_t)(b * S + k0 + 2 * rp) * KW + kvh * HD + d;
            v8h v0 = *(const v8h*)&Vh[off0];
            v8h v1 = *(const v8h*)&Vh[off0 + KW];
            #pragma unroll
            for (int j = 0; j < 8; ++j) {
                v2h p = { v0[j], v1[j] };
                *(v2h*)&Vt[d + j][2 * rp] = p;
            }
        }
        wait_async_zero();
        __syncthreads();

        // scores S = Q @ K^T  (16 x 64 per wave)
        v8f sc[4];
        #pragma unroll
        for (int nt = 0; nt < 4; ++nt) {
            v16h bk0 = load_b_frag(&Ks[nt * 16][0],  72);
            v16h bk1 = load_b_frag(&Ks[nt * 16][32], 72);
            v8f c = {};
            c = WMMA_F32_16x16x32_F16(aq0, bk0, c);
            c = WMMA_F32_16x16x32_F16(aq1, bk1, c);
            sc[nt] = c;
        }

        int kidx[4];
        #pragma unroll
        for (int nt = 0; nt < 4; ++nt) kidx[nt] = k0 + nt * 16 + nloc;

        // online softmax per row (fully unrolled: no dynamic VGPR indexing)
        #pragma unroll
        for (int r = 0; r < 8; ++r) {
            const int qidx = q0 + wq + mhi + r;
            float mx = -1e30f;
            #pragma unroll
            for (int nt = 0; nt < 4; ++nt) {
                float v = sc[nt][r] * scale;
                v = (kidx[nt] <= qidx) ? v : -1e30f;
                sc[nt][r] = v;
                mx = fmaxf(mx, v);
            }
            #pragma unroll
            for (int off = 1; off < 16; off <<= 1)       // 16-lane row reduce
                mx = fmaxf(mx, __shfl_xor(mx, off, 32));
            float mnew  = fmaxf(m_s[r], mx);
            float alpha = __expf(m_s[r] - mnew);
            float sum = 0.0f;
            #pragma unroll
            for (int nt = 0; nt < 4; ++nt) {
                float p = __expf(sc[nt][r] - mnew);
                sc[nt][r] = p;
                sum += p;
            }
            #pragma unroll
            for (int off = 1; off < 16; off <<= 1)
                sum += __shfl_xor(sum, off, 32);
            l_s[r] = l_s[r] * alpha + sum;
            m_s[r] = mnew;
            #pragma unroll
            for (int ht = 0; ht < 4; ++ht)
                acc[ht][r] *= alpha;
            // spill P row to wave-private LDS (D-layout -> row-major strip)
            #pragma unroll
            for (int nt = 0; nt < 4; ++nt)
                Ps[wave][mhi + r][nt * 16 + nloc] = (_Float16)sc[nt][r];
        }

        // O += P @ V   (A-frag reload of P; Vt is N-major over hd)
        v16h ap0 = load_a_frag(&Ps[wave][0][0],  72);
        v16h ap1 = load_a_frag(&Ps[wave][0][32], 72);
        #pragma unroll
        for (int ht = 0; ht < 4; ++ht) {
            v16h bv0 = load_b_frag(&Vt[ht * 16][0],  72);
            v16h bv1 = load_b_frag(&Vt[ht * 16][32], 72);
            acc[ht] = WMMA_F32_16x16x32_F16(ap0, bv0, acc[ht]);
            acc[ht] = WMMA_F32_16x16x32_F16(ap1, bv1, acc[ht]);
        }
        __syncthreads();
    }

    // normalize + write f16 attention output
    #pragma unroll
    for (int r = 0; r < 8; ++r) {
        const float inv_l = 1.0f / l_s[r];
        const int row = q0 + wq + mhi + r;
        #pragma unroll
        for (int ht = 0; ht < 4; ++ht) {
            int col = h * HD + ht * 16 + nloc;
            Oh[(size_t)(b * S + row) * QW + col] = (_Float16)(acc[ht][r] * inv_l);
        }
    }
}

// ---------------------------------------------------------------------------
// Host-side pipeline
// ---------------------------------------------------------------------------
extern "C" void kernel_launch(void* const* d_in, const int* in_sizes, int n_in,
                              void* d_out, int out_size, void* d_ws, size_t ws_size,
                              hipStream_t stream) {
    (void)in_sizes; (void)n_in; (void)out_size; (void)ws_size;
    const float* x    = (const float*)d_in[0];
    const float* cosb = (const float*)d_in[1];
    const float* sinb = (const float*)d_in[2];
    const float* Wq   = (const float*)d_in[3];
    const float* Wk   = (const float*)d_in[4];
    const float* Wv   = (const float*)d_in[5];
    const float* Wo   = (const float*)d_in[6];
    float* out        = (float*)d_out;

    const int Bv = 2, Sv = 2048, DIM = 2048, NHv = 32, NKVv = 8, HDv = 64;
    const int Mrows = Bv * Sv;                     // 4096
    const size_t QN = (size_t)Mrows * NHv * HDv;   // 8Mi halves
    const size_t KN = (size_t)Mrows * NKVv * HDv;  // 2Mi halves

    _Float16* Qh = (_Float16*)d_ws;
    _Float16* Kh = Qh + QN;
    _Float16* Vh = Kh + KN;
    _Float16* Oh = Vh + KN;

    // QKV projections (f32 -> f16 intermediates)
    gemm_wmma<float, _Float16><<<dim3((NHv*HDv)/BN,  Mrows/BM), 256, 0, stream>>>(x, Wq, Qh, Mrows, NHv*HDv,  DIM);
    gemm_wmma<float, _Float16><<<dim3((NKVv*HDv)/BN, Mrows/BM), 256, 0, stream>>>(x, Wk, Kh, Mrows, NKVv*HDv, DIM);
    gemm_wmma<float, _Float16><<<dim3((NKVv*HDv)/BN, Mrows/BM), 256, 0, stream>>>(x, Wv, Vh, Mrows, NKVv*HDv, DIM);

    // RoPE on Q and K (in place)
    int qpairs = Mrows * NHv * (HDv / 2);
    int kpairs = Mrows * NKVv * (HDv / 2);
    rope_kernel<<<(qpairs + 255) / 256, 256, 0, stream>>>(Qh, cosb, sinb, NHv,  qpairs);
    rope_kernel<<<(kpairs + 255) / 256, 256, 0, stream>>>(Kh, cosb, sinb, NKVv, kpairs);

    // causal flash attention
    flash_attn<<<dim3(Sv / 64, NHv, Bv), 128, 0, stream>>>(Qh, Kh, Vh, Oh);

    // output projection (f16 A, f32 out)
    gemm_wmma<_Float16, float><<<dim3(DIM/BN, Mrows/BM), 256, 0, stream>>>(Oh, Wo, out, Mrows, DIM, DIM);
}
